// LRUCell_80161269613378
// MI455X (gfx1250) — matrix operand order repaired and checked by reading, not compiled, not run on hardware
//
#include <hip/hip_runtime.h>
#include <hip/hip_bf16.h>

// ---------------------------------------------------------------------------
// LRU cell for MI455X (gfx1250, wave32, WMMA f16 16x16x32).
// Chunked parallel scan (C=256 chunks of L=8); all GEMMs via
// v_wmma_f32_16x16x32_f16; x tiles / carry state staged with
// global_load_async_to_lds_b128 (ASYNCcnt) and double-buffered so the DMA
// overlaps the WMMA work. hs/u never touch HBM (they would be 512MB).
// ---------------------------------------------------------------------------

typedef __attribute__((ext_vector_type(16))) _Float16 v16h;
typedef __attribute__((ext_vector_type(8)))  float    v8f;

static constexpr int T  = 2048;
static constexpr int Bb = 32;     // batch
static constexpr int I  = 256;    // input dim
static constexpr int H  = 512;    // hidden dim
static constexpr int O  = 256;    // output dim

static constexpr int L    = 8;    // chunk length (timesteps per chunk)
static constexpr int C    = 256;  // number of chunks  (C*L == T)
static constexpr int SUB  = 4;    // timesteps per sub-tile (N-tiles per GEMM)
static constexpr int NSUB = 2;    // sub-tiles per chunk  (NSUB*SUB == L)
static constexpr int BT   = 16;   // batch tile (one WMMA N dim)
static constexpr int NBT  = 2;    // batch tiles (NBT*BT == Bb)

// ---- LDS layout (dynamic shared, 268288 bytes < 320KB/WGP) ----------------
static constexpr int SOFF_UR = 0;        // SUB*BT*H f16 = 64KB   u_r / hs_r [t][b][h]
static constexpr int SOFF_UI = 65536;    // 64KB                  u_i / -hs_i
static constexpr int SOFF_X0 = 131072;   // 32KB  x tile, buffer 0
static constexpr int SOFF_X1 = 163840;   // 32KB  x tile, buffer 1
static constexpr int SOFF_FR = 196608;   // BT*H f32 = 32KB       h-state real
static constexpr int SOFF_FI = 229376;   // 32KB                  h-state imag
static constexpr int SOFF_G  = 262144;   // H f32 = 2KB           gamma
static constexpr int SOFF_LR = 264192;   // 2KB                   lam real
static constexpr int SOFF_LI = 266240;   // 2KB                   lam imag
static constexpr int SMEM_BYTES = 268288;

union AF  { v16h v; uint4 q[2]; };
union F8  { v8f  v; float4 q[2]; };
union H8  { uint4 q; _Float16 h[8]; };

__device__ __forceinline__ v8f wmma_f16(v16h a, v16h b, v8f c) {
  return __builtin_amdgcn_wmma_f32_16x16x32_f16(false, a, false, b, (short)0, c,
                                                false, false);
}

// ---- CDNA5 async global<->LDS helpers (ASYNCcnt) --------------------------
__device__ __forceinline__ unsigned lds_off(const void* p) {
  return (unsigned)(uintptr_t)p;   // flat LDS addr: addr[31:0] = LDS offset
}
__device__ __forceinline__ void async_g2l_b128(unsigned lds, const void* gbase,
                                               unsigned goff) {
  asm volatile("global_load_async_to_lds_b128 %0, %1, %2"
               :: "v"(lds), "v"(goff), "s"(gbase) : "memory");
}
__device__ __forceinline__ void async_l2g_b128(const void* gbase, unsigned goff,
                                               unsigned lds) {
  asm volatile("global_store_async_from_lds_b128 %0, %1, %2"
               :: "v"(goff), "v"(lds), "s"(gbase) : "memory");
}
__device__ __forceinline__ void wait_async0() {
#if __has_builtin(__builtin_amdgcn_s_wait_asynccnt)
  __builtin_amdgcn_s_wait_asynccnt(0);
#else
  asm volatile("s_wait_asynccnt 0x0" ::: "memory");
#endif
}

// Stage a [SUB t x BT b x I] f16 x-tile into LDS via async DMA (no wait here).
// 64 rows of 512B, 4 threads/row, 8x b128 per thread.
__device__ __forceinline__ void stage_x(const _Float16* __restrict__ xh,
                                        _Float16* sX, int t0, int bt, int tid) {
  const int r = tid >> 2, q = tid & 3;
  const int ts = r >> 4, b = r & 15;
  const unsigned goff = (unsigned)((((unsigned)(t0 + ts) * Bb + (bt * BT + b)) * I
                                    + q * 64) * 2u);
  const unsigned lds = lds_off(sX + (r * I + q * 64));
#pragma unroll
  for (int m = 0; m < 8; ++m)
    async_g2l_b128(lds + m * 16, xh, goff + m * 16);
  if (t0 + SUB < T)  // prefetch the tile after next (global_prefetch_b8)
    __builtin_prefetch((const char*)xh + goff + (size_t)SUB * Bb * I * 2, 0, 1);
}

// Input projection: u = gamma .* (W_ih @ x_t)  for SUB timesteps.
// D = [64 M-tiles (32 real + 32 imag) x 4 N-tiles], K = I = 8 wmma steps.
__device__ __forceinline__ void proj_gemm(const _Float16* __restrict__ wih_r,
                                          const _Float16* __restrict__ wih_i,
                                          const float* sG, const _Float16* sX,
                                          _Float16* sU_r, _Float16* sU_i, int tid) {
  const int wave = tid >> 5, lane = tid & 31;
  const int lrow = lane & 15, lhalf = lane >> 4;
#pragma unroll 1
  for (int j = 0; j < 8; ++j) {
    const int mt = wave * 8 + j;                 // 0..63
    const _Float16* W = (mt < 32) ? wih_r : wih_i;
    _Float16*      sU = (mt < 32) ? sU_r : sU_i;
    const int h16 = (mt & 31) * 16;
    const int row = h16 + lrow;
    AF a[8];                                     // cache A-frags across N-tiles
#pragma unroll
    for (int k = 0; k < 8; ++k) {
      const uint4* p = (const uint4*)(W + row * I + k * 32 + lhalf * 8);
      a[k].q[0] = p[0];                          // K = 8*half + 0..7
      a[k].q[1] = p[2];                          // K = 16 + 8*half + 0..7
    }
    float g[8];
#pragma unroll
    for (int r = 0; r < 8; ++r) g[r] = sG[h16 + lhalf * 8 + r];
#pragma unroll
    for (int nt = 0; nt < SUB; ++nt) {
      v8f acc = {};
#pragma unroll
      for (int k = 0; k < 8; ++k) {
        AF b;
        const uint4* bp = (const uint4*)(sX + ((nt * BT + lrow) * I + k * 32
                                               + lhalf * 8));
        b.q[0] = bp[0]; b.q[1] = bp[2];
        acc = wmma_f16(a[k].v, b.v, acc);
      }
      H8 out;
#pragma unroll
      for (int r = 0; r < 8; ++r) out.h[r] = (_Float16)(acc[r] * g[r]);
      *(uint4*)(sU + ((nt * BT + lrow) * H + h16 + lhalf * 8)) = out.q;
    }
  }
}

// ---------------------------------------------------------------------------
__global__ void k_setup(const float* __restrict__ nu, const float* __restrict__ th,
                        float* lam_r, float* lam_i, float* gam,
                        float* lamLr, float* lamLi) {
  const int h = threadIdx.x;
  if (h >= H) return;
  const float en = __expf(nu[h]);
  const float et = __expf(th[h]);
  const float mag = __expf(-en);
  const float lr = mag * __cosf(et), li = mag * __sinf(et);
  lam_r[h] = lr; lam_i[h] = li;
  gam[h] = sqrtf(fmaxf(0.f, 1.f - mag * mag));
  float ar = lr, ai = li;            // lam^L, L = 8 -> 3 squarings
#pragma unroll
  for (int s = 0; s < 3; ++s) { float nr = ar*ar - ai*ai, ni = 2.f*ar*ai; ar = nr; ai = ni; }
  lamLr[h] = ar; lamLi[h] = ai;
}

__global__ void k_cvt(const float* __restrict__ src, _Float16* __restrict__ dst, int n) {
  const int i = (blockIdx.x * blockDim.x + threadIdx.x) * 4;
  if (i + 3 < n) {
    const float4 v = *(const float4*)(src + i);
    union { _Float16 h[4]; uint2 u; } o;
    o.h[0] = (_Float16)v.x; o.h[1] = (_Float16)v.y;
    o.h[2] = (_Float16)v.z; o.h[3] = (_Float16)v.w;
    *(uint2*)(dst + i) = o.u;
  }
}

// Pass A: per-chunk zero-init local scan result ("finals").
__global__ void k_passA(const _Float16* __restrict__ xh,
                        const _Float16* __restrict__ wih_r,
                        const _Float16* __restrict__ wih_i,
                        const float* __restrict__ gam,
                        const float* __restrict__ lam_r,
                        const float* __restrict__ lam_i,
                        float* __restrict__ fin_r, float* __restrict__ fin_i) {
  extern __shared__ char smem[];
  _Float16* sU_r = (_Float16*)(smem + SOFF_UR);
  _Float16* sU_i = (_Float16*)(smem + SOFF_UI);
  _Float16* sX0  = (_Float16*)(smem + SOFF_X0);
  _Float16* sX1  = (_Float16*)(smem + SOFF_X1);
  float* sFr = (float*)(smem + SOFF_FR);
  float* sFi = (float*)(smem + SOFF_FI);
  float* sG  = (float*)(smem + SOFF_G);
  float* sLr = (float*)(smem + SOFF_LR);
  float* sLi = (float*)(smem + SOFF_LI);
  const int tid = threadIdx.x;
  const int bt = blockIdx.x, cy = blockIdx.y;

  stage_x(xh, sX0, cy * L, bt, tid);             // async x tile for sub 0
  for (int h = tid; h < H; h += 256) { sG[h] = gam[h]; sLr[h] = lam_r[h]; sLi[h] = lam_i[h]; }
  for (int j = tid; j < BT * H; j += 256) { sFr[j] = 0.f; sFi[j] = 0.f; }

  for (int s = 0; s < NSUB; ++s) {
    wait_async0();
    __syncthreads();                             // x[s] ready; sU free for reuse
    _Float16* sXc = (s & 1) ? sX1 : sX0;
    if (s + 1 < NSUB)                            // overlap next DMA with GEMMs
      stage_x(xh, (s & 1) ? sX0 : sX1, cy * L + (s + 1) * SUB, bt, tid);
    proj_gemm(wih_r, wih_i, sG, sXc, sU_r, sU_i, tid);
    __syncthreads();
#pragma unroll 1
    for (int jj = 0; jj < 32; ++jj) {            // local scan f = lam*f + u
      const int flat = tid + 256 * jj;
      const int b = flat >> 9, h = flat & 511;
      const float lr = sLr[h], li = sLi[h];
      float fr = sFr[flat], fi = sFi[flat];
#pragma unroll
      for (int ts = 0; ts < SUB; ++ts) {
        const int ui = (ts * BT + b) * H + h;
        const float ur = (float)sU_r[ui], uv = (float)sU_i[ui];
        const float nr = fmaf(lr, fr, fmaf(-li, fi, ur));
        const float ni = fmaf(lr, fi, fmaf(li, fr, uv));
        fr = nr; fi = ni;
      }
      sFr[flat] = fr; sFi[flat] = fi;
    }
    __syncthreads();
  }
  // finals -> global, async store straight from LDS (contiguous 2x32KB)
  const float* dst_r = fin_r + (size_t)cy * (Bb * H) + (size_t)bt * BT * H;
  const float* dst_i = fin_i + (size_t)cy * (Bb * H) + (size_t)bt * BT * H;
  const unsigned lr0 = lds_off(sFr), li0 = lds_off(sFi);
#pragma unroll
  for (int m = 0; m < 8; ++m) {
    const unsigned o = (unsigned)(m * 4096 + tid * 16);
    async_l2g_b128(dst_r, o, lr0 + o);
    async_l2g_b128(dst_i, o, li0 + o);
  }
  wait_async0();
}

// Pass B: sequential carry over chunks (in place: finals -> carry-in), emits h_last.
__global__ void k_carry(const float* __restrict__ hpr, const float* __restrict__ hpi,
                        const float* __restrict__ lamLr, const float* __restrict__ lamLi,
                        float* __restrict__ fin_r, float* __restrict__ fin_i,
                        float* __restrict__ out_hr, float* __restrict__ out_hi) {
  const int idx = blockIdx.x * blockDim.x + threadIdx.x;   // 0 .. B*H-1
  const int h = idx & (H - 1);
  float cr = hpr[idx], ci = hpi[idx];
  const float ar = lamLr[h], ai = lamLi[h];
#pragma unroll 1
  for (int c = 0; c < C; ++c) {
    const size_t o = (size_t)c * (Bb * H) + idx;
    const float fr = fin_r[o], fi = fin_i[o];
    fin_r[o] = cr; fin_i[o] = ci;                 // becomes carry-in for chunk c
    const float nr = fmaf(ar, cr, fmaf(-ai, ci, fr));
    const float ni = fmaf(ar, ci, fmaf(ai, cr, fi));
    cr = nr; ci = ni;
  }
  out_hr[idx] = cr; out_hi[idx] = ci;             // h_last
}

// Pass C: replay chunk from true carry-in, fused output GEMM over 4x16 N-tiles.
__global__ void k_passC(const _Float16* __restrict__ xh,
                        const _Float16* __restrict__ wih_r,
                        const _Float16* __restrict__ wih_i,
                        const _Float16* __restrict__ who_r,
                        const _Float16* __restrict__ who_i,
                        const _Float16* __restrict__ wio,
                        const float* __restrict__ gam,
                        const float* __restrict__ lam_r,
                        const float* __restrict__ lam_i,
                        const float* __restrict__ fin_r,
                        const float* __restrict__ fin_i,
                        float* __restrict__ y) {
  extern __shared__ char smem[];
  _Float16* sU_r = (_Float16*)(smem + SOFF_UR);
  _Float16* sU_i = (_Float16*)(smem + SOFF_UI);
  _Float16* sX0  = (_Float16*)(smem + SOFF_X0);
  _Float16* sX1  = (_Float16*)(smem + SOFF_X1);
  float* sFr = (float*)(smem + SOFF_FR);
  float* sFi = (float*)(smem + SOFF_FI);
  float* sG  = (float*)(smem + SOFF_G);
  float* sLr = (float*)(smem + SOFF_LR);
  float* sLi = (float*)(smem + SOFF_LI);
  const int tid = threadIdx.x;
  const int bt = blockIdx.x, cy = blockIdx.y;
  const int wave = tid >> 5, lane = tid & 31;
  const int lrow = lane & 15, lhalf = lane >> 4;

  // async: carry-in h state (2x32KB, contiguous) + x tile for sub 0
  const float* src_r = fin_r + (size_t)cy * (Bb * H) + (size_t)bt * BT * H;
  const float* src_i = fin_i + (size_t)cy * (Bb * H) + (size_t)bt * BT * H;
  {
    const unsigned lr0 = lds_off(sFr), li0 = lds_off(sFi);
#pragma unroll
    for (int m = 0; m < 8; ++m) {
      const unsigned o = (unsigned)(m * 4096 + tid * 16);
      async_g2l_b128(lr0 + o, src_r, o);
      async_g2l_b128(li0 + o, src_i, o);
    }
  }
  stage_x(xh, sX0, cy * L, bt, tid);
  for (int h = tid; h < H; h += 256) { sG[h] = gam[h]; sLr[h] = lam_r[h]; sLi[h] = lam_i[h]; }

  for (int s = 0; s < NSUB; ++s) {
    wait_async0();
    __syncthreads();                             // x[s] + carry ready; sU free
    _Float16* sXc = (s & 1) ? sX1 : sX0;
    if (s + 1 < NSUB)                            // overlap next DMA with GEMMs
      stage_x(xh, (s & 1) ? sX0 : sX1, cy * L + (s + 1) * SUB, bt, tid);
    const int t0 = cy * L + s * SUB;
    proj_gemm(wih_r, wih_i, sG, sXc, sU_r, sU_i, tid);
    __syncthreads();
#pragma unroll 1
    for (int jj = 0; jj < 32; ++jj) {             // scan; write hs back into U
      const int flat = tid + 256 * jj;
      const int b = flat >> 9, h = flat & 511;
      const float lr = sLr[h], li = sLi[h];
      float hr = sFr[flat], hi = sFi[flat];
#pragma unroll
      for (int ts = 0; ts < SUB; ++ts) {
        const int ui = (ts * BT + b) * H + h;
        const float ur = (float)sU_r[ui], uv = (float)sU_i[ui];
        const float nr = fmaf(lr, hr, fmaf(-li, hi, ur));
        const float ni = fmaf(lr, hi, fmaf(li, hr, uv));
        hr = nr; hi = ni;
        sU_r[ui] = (_Float16)hr;
        sU_i[ui] = (_Float16)(-hi);               // fold the minus of Re(C@h)
      }
      sFr[flat] = hr; sFi[flat] = hi;
    }
    __syncthreads();

    // Output GEMM: y = who_r@hs_r + who_i@(-hs_i) + wio@x ; M=O (16 tiles).
    const int bg = bt * BT + lrow;
#pragma unroll 1
    for (int j = 0; j < 2; ++j) {
      const int mt = wave * 2 + j;
      const int orow = mt * 16 + lrow;
      v8f acc[SUB];
#pragma unroll
      for (int nt = 0; nt < SUB; ++nt) acc[nt] = (v8f){};
#pragma unroll 1
      for (int k = 0; k < 16; ++k) {              // K = H, real part
        AF a;
        const uint4* p = (const uint4*)(who_r + orow * H + k * 32 + lhalf * 8);
        a.q[0] = p[0]; a.q[1] = p[2];
#pragma unroll
        for (int nt = 0; nt < SUB; ++nt) {
          AF b;
          const uint4* bp = (const uint4*)(sU_r + ((nt * BT + lrow) * H + k * 32
                                                   + lhalf * 8));
          b.q[0] = bp[0]; b.q[1] = bp[2];
          acc[nt] = wmma_f16(a.v, b.v, acc[nt]);
        }
      }
#pragma unroll 1
      for (int k = 0; k < 16; ++k) {              // K = H, (-imag) part
        AF a;
        const uint4* p = (const uint4*)(who_i + orow * H + k * 32 + lhalf * 8);
        a.q[0] = p[0]; a.q[1] = p[2];
#pragma unroll
        for (int nt = 0; nt < SUB; ++nt) {
          AF b;
          const uint4* bp = (const uint4*)(sU_i + ((nt * BT + lrow) * H + k * 32
                                                   + lhalf * 8));
          b.q[0] = bp[0]; b.q[1] = bp[2];
          acc[nt] = wmma_f16(a.v, b.v, acc[nt]);
        }
      }
#pragma unroll 1
      for (int k = 0; k < 8; ++k) {               // K = I, D @ x term
        AF a;
        const uint4* p = (const uint4*)(wio + orow * I + k * 32 + lhalf * 8);
        a.q[0] = p[0]; a.q[1] = p[2];
#pragma unroll
        for (int nt = 0; nt < SUB; ++nt) {
          AF b;
          const uint4* bp = (const uint4*)(sXc + ((nt * BT + lrow) * I + k * 32
                                                  + lhalf * 8));
          b.q[0] = bp[0]; b.q[1] = bp[2];
          acc[nt] = wmma_f16(a.v, b.v, acc[nt]);
        }
      }
#pragma unroll
      for (int nt = 0; nt < SUB; ++nt) {
        const int tg = t0 + nt;
        float* dst = y + ((size_t)tg * Bb + bg) * O + mt * 16 + lhalf * 8;
        F8 st; st.v = acc[nt];
        *(float4*)dst = st.q[0];
        *(float4*)(dst + 4) = st.q[1];
      }
    }
  }
}

// ---------------------------------------------------------------------------
extern "C" void kernel_launch(void* const* d_in, const int* in_sizes, int n_in,
                              void* d_out, int out_size, void* d_ws, size_t ws_size,
                              hipStream_t stream) {
  (void)in_sizes; (void)n_in; (void)out_size; (void)ws_size;
  const float* x     = (const float*)d_in[0];
  const float* hpr   = (const float*)d_in[1];
  const float* hpi   = (const float*)d_in[2];
  const float* wihr  = (const float*)d_in[3];
  const float* wihi  = (const float*)d_in[4];
  const float* nu    = (const float*)d_in[5];
  const float* theta = (const float*)d_in[6];
  const float* whor  = (const float*)d_in[7];
  const float* whoi  = (const float*)d_in[8];
  const float* wiof  = (const float*)d_in[9];

  char* ws = (char*)d_ws;
  float* lam_r = (float*)ws;                 // 5*H f32
  float* lam_i = lam_r + H;
  float* gam   = lam_i + H;
  float* lamLr = gam + H;
  float* lamLi = lamLr + H;
  _Float16* wih_r = (_Float16*)(ws + 16384); // f16 weights: 1.15MB total
  _Float16* wih_i = wih_r + H * I;
  _Float16* who_r = wih_i + H * I;
  _Float16* who_i = who_r + O * H;
  _Float16* wio   = who_i + O * H;
  _Float16* xh    = (_Float16*)(ws + (size_t)(2 << 20));            // 32MB
  float* fin_r    = (float*)(ws + (size_t)(2 << 20) + (size_t)T * Bb * I * 2);
  float* fin_i    = fin_r + (size_t)C * Bb * H;                      // 2x16MB

  float* y      = (float*)d_out;
  float* out_hr = y + (size_t)T * Bb * O;
  float* out_hi = out_hr + Bb * H;

  k_setup<<<1, 512, 0, stream>>>(nu, theta, lam_r, lam_i, gam, lamLr, lamLi);
  k_cvt<<<(T * Bb * I / 4 + 255) / 256, 256, 0, stream>>>(x, xh, T * Bb * I);
  k_cvt<<<(H * I / 4 + 255) / 256, 256, 0, stream>>>(wihr, wih_r, H * I);
  k_cvt<<<(H * I / 4 + 255) / 256, 256, 0, stream>>>(wihi, wih_i, H * I);
  k_cvt<<<(O * H / 4 + 255) / 256, 256, 0, stream>>>(whor, who_r, O * H);
  k_cvt<<<(O * H / 4 + 255) / 256, 256, 0, stream>>>(whoi, who_i, O * H);
  k_cvt<<<(O * I / 4 + 255) / 256, 256, 0, stream>>>(wiof, wio, O * I);

  const dim3 grid(NBT, C);
  k_passA<<<grid, 256, SMEM_BYTES, stream>>>(xh, wih_r, wih_i, gam, lam_r, lam_i,
                                             fin_r, fin_i);
  k_carry<<<(Bb * H) / 256, 256, 0, stream>>>(hpr, hpi, lamLr, lamLi,
                                              fin_r, fin_i, out_hr, out_hi);
  k_passC<<<grid, 256, SMEM_BYTES, stream>>>(xh, wih_r, wih_i, who_r, who_i, wio,
                                             gam, lam_r, lam_i, fin_r, fin_i, y);
}
